// DynamicSparseAttention_37641093382283
// MI455X (gfx1250) — matrix operand order repaired and checked by reading, not compiled, not run on hardware
//
#include <hip/hip_runtime.h>

// ---------------------------------------------------------------------------
// Types for CDNA5 WMMA (wave32)
// ---------------------------------------------------------------------------
typedef __attribute__((ext_vector_type(16))) __bf16       v16bf;
typedef __attribute__((ext_vector_type(8)))  float        v8f;
typedef __attribute__((ext_vector_type(8)))  unsigned int v8u;

#define EMBED 512
#define SEQL  2048
#define NBATCH 4
#define HEADS 8
#define HDIM  64
#define TOPK  204   // int((1.0-0.9)*2048)

// GEMM block tiling
#define BM 128
#define BN 128
#define BK 32

// ---------------------------------------------------------------------------
// Async global->LDS copy (CDNA5 GLOBAL_LOAD_ASYNC_TO_LDS_B128, ASYNCcnt).
// The builtin takes typed pointers: (v4i addrspace(1)*, v4i addrspace(3)*,
// imm offset, imm cpol) — per hipcc's diagnostic on the previous round.
// ---------------------------------------------------------------------------
#if defined(__gfx1250__) &&                                              \
    __has_builtin(__builtin_amdgcn_global_load_async_to_lds_b128) &&     \
    __has_builtin(__builtin_amdgcn_s_wait_asynccnt)
#define HAVE_ASYNC_LDS 1
#else
#define HAVE_ASYNC_LDS 0
#endif

typedef int v4i_t __attribute__((vector_size(16)));
#if HAVE_ASYNC_LDS
typedef __attribute__((address_space(1))) v4i_t as1_v4i;
typedef __attribute__((address_space(3))) v4i_t as3_v4i;
#endif

__device__ __forceinline__ void cp_b128(void* lds_dst, const void* gsrc) {
#if HAVE_ASYNC_LDS
  __builtin_amdgcn_global_load_async_to_lds_b128(
      (as1_v4i*)gsrc, (as3_v4i*)lds_dst, 0, 0);
#else
  *(uint4*)lds_dst = *(const uint4*)gsrc;
#endif
}

__device__ __forceinline__ void cp_fence() {
#if HAVE_ASYNC_LDS
  __builtin_amdgcn_s_wait_asynccnt(0);
#endif
}

// ---------------------------------------------------------------------------
// Helpers
// ---------------------------------------------------------------------------
__device__ __forceinline__ unsigned short f32_to_bf16(float f) {
  unsigned int u = __float_as_uint(f);
  u += 0x7FFFu + ((u >> 16) & 1u);   // round-to-nearest-even
  return (unsigned short)(u >> 16);
}

__device__ __forceinline__ v16bf pack16(uint4 x, uint4 y) {
  v8u v;
  v[0] = x.x; v[1] = x.y; v[2] = x.z; v[3] = x.w;
  v[4] = y.x; v[5] = y.y; v[6] = y.z; v[7] = y.w;
  return __builtin_bit_cast(v16bf, v);
}

// A-matrix tile (16x32 bf16, MxK) from a row-major [M][ld] source.
// lane<16 -> M=lane, K in {0..7,16..23}; lane>=16 -> M=lane-16, K in {8..15,24..31}.
__device__ __forceinline__ v16bf load_a_tile(const unsigned short* base, int ld) {
  int lane = threadIdx.x & 31;
  int r = lane & 15;
  int c = (lane & 16) ? 8 : 0;
  const unsigned short* p = base + (size_t)r * ld + c;
  return pack16(*(const uint4*)p, *(const uint4*)(p + 16));
}

// B-matrix tile (32x16 bf16, KxN); memory source row-major [N][ld] (B = src^T).
// lane<16 -> N=lane, K in {0..7,16..23}; lane>=16 -> N=lane-16, K in {8..15,24..31}.
__device__ __forceinline__ v16bf load_b_tile(const unsigned short* base, int ld) {
  int lane = threadIdx.x & 31;
  int r = lane & 15;
  int c = (lane & 16) ? 8 : 0;
  const unsigned short* p = base + (size_t)r * ld + c;
  return pack16(*(const uint4*)p, *(const uint4*)(p + 16));
}

__device__ __forceinline__ v8f wmma_bf16(v16bf a, v16bf b, v8f c) {
  return __builtin_amdgcn_wmma_f32_16x16x32_bf16(false, a, false, b,
                                                 (short)0, c, false, false);
}

// ---------------------------------------------------------------------------
// f32 -> bf16 conversion
// ---------------------------------------------------------------------------
__global__ __launch_bounds__(256)
void cvt_bf16_kernel(const float* __restrict__ src,
                     unsigned short* __restrict__ dst, int n) {
  int i = blockIdx.x * blockDim.x + threadIdx.x;
  int stride = gridDim.x * blockDim.x;
  for (; i < n; i += stride) dst[i] = f32_to_bf16(src[i]);
}

// ---------------------------------------------------------------------------
// D = A(MxK) @ W^T (W is [N][K] row-major), bf16 in, f32 accumulate.
// LDS-staged 128x128 block per workgroup; async panel copies; 8 waves in a
// 4x2 grid, each computing a 32x64 register block (2x4 WMMA tiles).
// mode 0: bf16 [M][N]; mode 1: bf16 transposed per-batch (V^T);
// mode 2: f32 [M][N] + bias.
// ---------------------------------------------------------------------------
__global__ __launch_bounds__(256)
void gemm_bf16_kernel(const unsigned short* __restrict__ A,
                      const unsigned short* __restrict__ W,
                      unsigned short* __restrict__ Dbf,
                      float* __restrict__ Df,
                      const float* __restrict__ bias,
                      int M, int N, int K, int mode) {
  __shared__ unsigned short sA[BM * BK];   // 8 KB
  __shared__ unsigned short sB[BN * BK];   // 8 KB

  const int wave = threadIdx.x >> 5;
  const int lane = threadIdx.x & 31;
  const int nbn  = N / BN;
  const int bm   = (int)(blockIdx.x / nbn) * BM;
  const int bn   = (int)(blockIdx.x % nbn) * BN;

  const int wm = (wave >> 1) * 32;   // 0,32,64,96
  const int wn = (wave & 1) * 64;    // 0,64

  const int crow = threadIdx.x >> 1;        // 0..127
  const int ccol = (threadIdx.x & 1) * 16;  // 0 or 16

  v8f acc[2][4] = {};

  for (int k0 = 0; k0 < K; k0 += BK) {
    const unsigned short* ga = A + (size_t)(bm + crow) * K + k0 + ccol;
    const unsigned short* gb = W + (size_t)(bn + crow) * K + k0 + ccol;
    cp_b128(sA + crow * BK + ccol,     ga);
    cp_b128(sA + crow * BK + ccol + 8, ga + 8);
    cp_b128(sB + crow * BK + ccol,     gb);
    cp_b128(sB + crow * BK + ccol + 8, gb + 8);
    cp_fence();
    __syncthreads();

    v16bf a0 = load_a_tile(sA + (wm + 0)  * BK, BK);
    v16bf a1 = load_a_tile(sA + (wm + 16) * BK, BK);
#pragma unroll
    for (int j = 0; j < 4; j++) {
      v16bf b = load_b_tile(sB + (wn + j * 16) * BK, BK);
      acc[0][j] = wmma_bf16(a0, b, acc[0][j]);
      acc[1][j] = wmma_bf16(a1, b, acc[1][j]);
    }
    __syncthreads();
  }

#pragma unroll
  for (int i2 = 0; i2 < 2; i2++) {
#pragma unroll
    for (int j = 0; j < 4; j++) {
      int tmt = bm + wm + i2 * 16;
      int tnt = bn + wn + j * 16;
      int nn  = tnt + (lane & 15);
      int mb  = tmt + ((lane >> 4) << 3);
      if (mode == 2) {
        float bv = bias[nn];
#pragma unroll
        for (int i = 0; i < 8; i++)
          Df[(size_t)(mb + i) * N + nn] = acc[i2][j][i] + bv;
      } else if (mode == 1) {
#pragma unroll
        for (int i = 0; i < 8; i++) {
          int m  = mb + i;
          int b  = m >> 11;        // /2048
          int kp = m & 2047;
          Dbf[((size_t)b * N + nn) * SEQL + kp] = f32_to_bf16(acc[i2][j][i]);
        }
      } else {
#pragma unroll
        for (int i = 0; i < 8; i++)
          Dbf[(size_t)(mb + i) * N + nn] = f32_to_bf16(acc[i2][j][i]);
      }
    }
  }
}

// ---------------------------------------------------------------------------
// Fused sparse attention for one (n, h, 16-query-row block).
// Dynamic LDS: 16x2048 f32 energies (128KB) + 16x2048 bf16 weights (64KB).
// ---------------------------------------------------------------------------
__global__ __launch_bounds__(256)
void sparse_attn_kernel(const unsigned short* __restrict__ qp,
                        const unsigned short* __restrict__ kp,
                        const unsigned short* __restrict__ Vt,
                        unsigned short* __restrict__ attn_out) {
  extern __shared__ char smem[];
  float* sE = (float*)smem;                                     // [16][2048]
  unsigned short* sW = (unsigned short*)(smem + 16 * SEQL * 4); // [16][2048]

  const int L = SEQL, E = EMBED;
  int bid   = blockIdx.x;
  int qtile = bid & 127;
  int h     = (bid >> 7) & 7;
  int n     = bid >> 10;
  int qbase = qtile << 4;
  int wave  = threadIdx.x >> 5;
  int lane  = threadIdx.x & 31;

  const unsigned short* Qb = qp + ((size_t)(n * L + qbase)) * E + h * HDIM;
  const unsigned short* Kb = kp + (size_t)n * L * E + h * HDIM;

  // ---- Phase A: energy = Q_tile @ K^T (16 x 2048) into LDS --------------
  v16bf a0 = load_a_tile(Qb, E);        // d = 0..31
  v16bf a1 = load_a_tile(Qb + 32, E);   // d = 32..63
  for (int t = wave; t < (L >> 4); t += 8) {
    const unsigned short* kt = Kb + (size_t)(t << 4) * E;
    v16bf b0 = load_b_tile(kt, E);
    v16bf b1 = load_b_tile(kt + 32, E);
    v8f c = {};
    c = wmma_bf16(a0, b0, c);
    c = wmma_bf16(a1, b1, c);
    int col = (t << 4) + (lane & 15);
    int m0  = (lane >> 4) << 3;
#pragma unroll
    for (int i = 0; i < 8; i++) sE[(m0 + i) * L + col] = c[i];
  }
  __syncthreads();

  // ---- Phase B: softmax + exact top-k threshold + sparsify --------------
  for (int rr = 0; rr < 2; rr++) {
    int row = wave * 2 + rr;
    float* e = sE + (size_t)row * L;
    float m = -3.0e38f;
    for (int i = lane; i < L; i += 32) m = fmaxf(m, e[i]);
    for (int o = 16; o > 0; o >>= 1) m = fmaxf(m, __shfl_xor(m, o, 32));
    float s = 0.f;
    for (int i = lane; i < L; i += 32) {
      float v = __expf(e[i] - m);
      e[i] = v;
      s += v;
    }
    for (int o = 16; o > 0; o >>= 1) s += __shfl_xor(s, o, 32);

    // kth-largest (k=TOPK) via bit-pattern binary search:
    // max bits b s.t. count(x >= float(b)) >= TOPK  ->  b == bits(v_k).
    unsigned int lo = 0u, hi = 0x3F800000u;  // exp(x-max) in (0,1]
    while (lo < hi) {
      unsigned int mid = (lo + hi + 1u) >> 1;
      float t = __uint_as_float(mid);
      int cnt = 0;
      for (int i = lane; i < L; i += 32) cnt += (e[i] >= t) ? 1 : 0;
      for (int o = 16; o > 0; o >>= 1) cnt += __shfl_xor(cnt, o, 32);
      if (cnt >= TOPK) lo = mid; else hi = mid - 1u;
    }
    float thr = __uint_as_float(lo);
    float inv = 1.f / s;
    unsigned short* w = sW + (size_t)row * L;
    for (int i = lane; i < L; i += 32) {
      float v = e[i];
      w[i] = f32_to_bf16((v >= thr) ? v * inv : 0.f);
    }
  }
  __syncthreads();

  // ---- Phase C: out = sparse_attn(16x2048) @ V(2048x64) -----------------
  if (wave < 4) {
    const unsigned short* Vb =
        Vt + ((size_t)n * E + h * HDIM + wave * 16) * L;
    v8f c = {};
    for (int k = 0; k < L; k += 32) {
      v16bf a = load_a_tile(sW + k, L);
      v16bf b = load_b_tile(Vb + k, L);
      c = wmma_bf16(a, b, c);
    }
    int nn = h * HDIM + wave * 16 + (lane & 15);
    int m0 = qbase + ((lane >> 4) << 3);
#pragma unroll
    for (int i = 0; i < 8; i++)
      attn_out[(size_t)(n * L + m0 + i) * E + nn] = f32_to_bf16(c[i]);
  }
}

// ---------------------------------------------------------------------------
// Host launch
// ---------------------------------------------------------------------------
extern "C" void kernel_launch(void* const* d_in, const int* in_sizes, int n_in,
                              void* d_out, int out_size, void* d_ws, size_t ws_size,
                              hipStream_t stream) {
  (void)in_sizes; (void)n_in; (void)out_size; (void)ws_size;
  const float* values = (const float*)d_in[0];
  const float* keysf  = (const float*)d_in[1];
  const float* query  = (const float*)d_in[2];
  const float* Wv     = (const float*)d_in[3];
  const float* Wk     = (const float*)d_in[4];
  const float* Wq     = (const float*)d_in[5];
  const float* Wo     = (const float*)d_in[6];
  const float* bo     = (const float*)d_in[7];
  float* out = (float*)d_out;

  const int NL  = NBATCH * SEQL;          // 8192
  const int XSZ = NL * EMBED;             // 4194304
  const int WSZ = EMBED * EMBED;          // 262144

  unsigned short* ws = (unsigned short*)d_ws;
  unsigned short* vb  = ws;
  unsigned short* kb  = vb  + XSZ;
  unsigned short* qb  = kb  + XSZ;
  unsigned short* Wvb = qb  + XSZ;
  unsigned short* Wkb = Wvb + WSZ;
  unsigned short* Wqb = Wkb + WSZ;
  unsigned short* Wob = Wqb + WSZ;
  unsigned short* qp  = Wob + WSZ;
  unsigned short* kp  = qp  + XSZ;
  unsigned short* Vt  = kp  + XSZ;
  unsigned short* ao  = Vt  + XSZ;

  cvt_bf16_kernel<<<2048, 256, 0, stream>>>(values, vb, XSZ);
  cvt_bf16_kernel<<<2048, 256, 0, stream>>>(keysf,  kb, XSZ);
  cvt_bf16_kernel<<<2048, 256, 0, stream>>>(query,  qb, XSZ);
  cvt_bf16_kernel<<<256,  256, 0, stream>>>(Wv, Wvb, WSZ);
  cvt_bf16_kernel<<<256,  256, 0, stream>>>(Wk, Wkb, WSZ);
  cvt_bf16_kernel<<<256,  256, 0, stream>>>(Wq, Wqb, WSZ);
  cvt_bf16_kernel<<<256,  256, 0, stream>>>(Wo, Wob, WSZ);

  const int gblocks = (NL / BM) * (EMBED / BN);   // 64*4 = 256
  gemm_bf16_kernel<<<gblocks, 256, 0, stream>>>(qb, Wqb, qp, nullptr, nullptr,
                                                NL, EMBED, EMBED, 0);
  gemm_bf16_kernel<<<gblocks, 256, 0, stream>>>(kb, Wkb, kp, nullptr, nullptr,
                                                NL, EMBED, EMBED, 0);
  gemm_bf16_kernel<<<gblocks, 256, 0, stream>>>(vb, Wvb, Vt, nullptr, nullptr,
                                                NL, EMBED, EMBED, 1);

  const size_t lds_bytes = (size_t)16 * SEQL * 4 + (size_t)16 * SEQL * 2; // 192KB
  sparse_attn_kernel<<<4096, 256, lds_bytes, stream>>>(qp, kp, Vt, ao);

  gemm_bf16_kernel<<<gblocks, 256, 0, stream>>>(ao, Wob, nullptr, out, bo,
                                                NL, EMBED, EMBED, 2);
}
